// TransformerBlock_55456617726385
// MI455X (gfx1250) — matrix operand order repaired and checked by reading, not compiled, not run on hardware
//
#include <hip/hip_runtime.h>

// ---------------------------------------------------------------------------
// Types / helpers
// ---------------------------------------------------------------------------
typedef __attribute__((ext_vector_type(16))) unsigned short v16u;
typedef __attribute__((ext_vector_type(16))) __bf16         v16bf;
typedef __attribute__((ext_vector_type(8)))  float          v8f;
typedef __attribute__((ext_vector_type(4)))  int            v4i;

// Hardware bf16 converts (v_cvt_*bf16*) instead of software RNE bit-twiddling.
__device__ __forceinline__ unsigned short f2bf(float f) {
    __bf16 h = (__bf16)f;
    return __builtin_bit_cast(unsigned short, h);
}
__device__ __forceinline__ float bf2f(unsigned short u) {
    __bf16 h = __builtin_bit_cast(__bf16, u);
    return (float)h;
}
__device__ __forceinline__ v8f wmma_bf16(v16u a, v16u b, v8f c) {
    // v_wmma_f32_16x16x32_bf16 (neg_a, A, neg_b, B, c_mod, C, reuse_a, reuse_b)
    return __builtin_amdgcn_wmma_f32_16x16x32_bf16(
        false, __builtin_bit_cast(v16bf, a),
        false, __builtin_bit_cast(v16bf, b),
        (short)0, c, false, false);
}

// ---------------------------------------------------------------------------
// CDNA5 async global->LDS copy (GLOBAL_LOAD_ASYNC_TO_LDS_B128, ASYNCcnt),
// with a synchronous fallback if the builtin is not declared.
// Probe-derived prototype: arg0 = global int4* (AS1), arg1 = LDS int4* (AS3),
// then immediate offset and cpol.
// ---------------------------------------------------------------------------
#if defined(__AMDGCN__) && __has_builtin(__builtin_amdgcn_global_load_async_to_lds_b128)
#define HAVE_ASYNC_LDS 1
typedef __attribute__((address_space(1))) v4i g_v4i;
typedef __attribute__((address_space(3))) v4i l_v4i;
#else
#define HAVE_ASYNC_LDS 0
#endif

__device__ __forceinline__ void copy16(void* lds_dst, const void* g_src) {
#if HAVE_ASYNC_LDS
    __builtin_amdgcn_global_load_async_to_lds_b128(
        (g_v4i*)g_src, (l_v4i*)lds_dst, 0, 0);
#else
    unsigned long long* d = (unsigned long long*)lds_dst;
    const unsigned long long* s = (const unsigned long long*)g_src;
    d[0] = s[0]; d[1] = s[1];
#endif
}
__device__ __forceinline__ void async_wait0() {
#if HAVE_ASYNC_LDS
#if __has_builtin(__builtin_amdgcn_s_wait_asynccnt)
    __builtin_amdgcn_s_wait_asynccnt(0);
#else
    asm volatile("s_wait_asynccnt 0" ::: "memory");
#endif
#endif
}

#define DIMC   768
#define NHEADS 12
#define HDIM   64
#define HIDDEN 1536
#define SEQ    1024
#define BATCH  8
#define MROWS  (BATCH * SEQ)   // 8192

// ---------------------------------------------------------------------------
// f32 -> bf16 elementwise (weights)
// ---------------------------------------------------------------------------
__global__ void cvt_f32_bf16(const float* __restrict__ in,
                             unsigned short* __restrict__ out, int n) {
    int i = blockIdx.x * blockDim.x + threadIdx.x;
    if (i < n) out[i] = f2bf(in[i]);
}

// ---------------------------------------------------------------------------
// LayerNorm over rows of 768, output bf16
// ---------------------------------------------------------------------------
__global__ __launch_bounds__(256)
void ln_kernel(const float* __restrict__ x, const float* __restrict__ g,
               const float* __restrict__ bb, unsigned short* __restrict__ out) {
    __shared__ float r1[256];
    __shared__ float r2[256];
    const int row = blockIdx.x;
    const int t   = threadIdx.x;
    const float* xr = x + (size_t)row * DIMC;
    float v0 = xr[t], v1 = xr[t + 256], v2 = xr[t + 512];
    r1[t] = v0 + v1 + v2;
    r2[t] = v0 * v0 + v1 * v1 + v2 * v2;
    __syncthreads();
    for (int off = 128; off > 0; off >>= 1) {
        if (t < off) { r1[t] += r1[t + off]; r2[t] += r2[t + off]; }
        __syncthreads();
    }
    const float mean = r1[0] * (1.0f / DIMC);
    const float var  = r2[0] * (1.0f / DIMC) - mean * mean;
    const float rstd = rsqrtf(var + 1e-5f);
    unsigned short* orow = out + (size_t)row * DIMC;
    orow[t]       = f2bf((v0 - mean) * rstd * g[t]       + bb[t]);
    orow[t + 256] = f2bf((v1 - mean) * rstd * g[t + 256] + bb[t + 256]);
    orow[t + 512] = f2bf((v2 - mean) * rstd * g[t + 512] + bb[t + 512]);
}

// ---------------------------------------------------------------------------
// Generic bf16 GEMM, f32 accumulate.  C[M,N] = A[M,K] * B[K,N] (+bias)(+gelu)(+res)
// 256 threads = 8 waves, tile 128x64, k-step 32. Wave w -> rows [16w,16w+16),
// 4 accumulators (16x16) across the 64 output cols. Tiles staged via async LDS.
// ---------------------------------------------------------------------------
template <bool BIAS, bool RES, bool GELU_ACT, bool OUT_BF16>
__global__ __launch_bounds__(256)
void gemm_bf16_kernel(const unsigned short* __restrict__ A,
                      const unsigned short* __restrict__ Bm,
                      const float* __restrict__ bias,
                      const float* __restrict__ res,
                      float* __restrict__ outF,
                      unsigned short* __restrict__ outB,
                      int M, int N, int K) {
    constexpr int BM = 128, BN = 64, BK = 32;
    constexpr int SAP = 40;   // pitches: 16B-aligned rows + bank-conflict break
    constexpr int SBP = 72;
    __shared__ unsigned short sA[BM * SAP];
    __shared__ unsigned short sB[BK * SBP];

    const int tid  = threadIdx.x;
    const int wave = tid >> 5;
    const int lane = tid & 31;
    const int rowBase = blockIdx.x * BM;
    const int colBase = blockIdx.y * BN;

    v8f acc[4] = {};

    const int ar   = (wave << 4) + (lane & 15);   // A-frag row within tile
    const int kb   = (lane >> 4) << 3;            // A-frag k base (0 or 8)
    const int koff = (lane >> 4) << 4;            // unused; B-frag also uses kb
    const int bn   = lane & 15;                   // B-frag column

    for (int k0 = 0; k0 < K; k0 += BK) {
        // stage A 128x32 (512 x 16B) and B 32x64 (256 x 16B)
        for (int i = tid; i < 512; i += 256) {
            int r = i >> 2, c = (i & 3) << 3;
            copy16(&sA[r * SAP + c], &A[(size_t)(rowBase + r) * K + k0 + c]);
        }
        {
            int i = tid;  // exactly 256 chunks
            int r = i >> 3, c = (i & 7) << 3;
            copy16(&sB[r * SBP + c], &Bm[(size_t)(k0 + r) * N + colBase + c]);
        }
        async_wait0();
        __syncthreads();

        v16u afrag;
#pragma unroll
        for (int e = 0; e < 8; ++e) {
            afrag[e]     = sA[ar * SAP + kb + e];
            afrag[e + 8] = sA[ar * SAP + kb + 16 + e];
        }
#pragma unroll
        for (int c4 = 0; c4 < 4; ++c4) {
            v16u bfrag;
#pragma unroll
            for (int e = 0; e < 16; ++e)
                bfrag[e] = sB[(kb + e + ((e >> 3) << 3)) * SBP + (c4 << 4) + bn];
            acc[c4] = wmma_bf16(afrag, bfrag, acc[c4]);
        }
        __syncthreads();
    }

    // epilogue: C layout VGPR i -> row i (lanes 0-15) / i+8 (lanes 16-31)
    const int rloc = (wave << 4) + ((lane >> 4) << 3);
    const int cl   = lane & 15;
#pragma unroll
    for (int c4 = 0; c4 < 4; ++c4) {
#pragma unroll
        for (int i = 0; i < 8; ++i) {
            const int grow = rowBase + rloc + i;
            const int gcol = colBase + (c4 << 4) + cl;
            float v = acc[c4][i];
            if (BIAS)     v += bias[gcol];
            if (GELU_ACT) v  = 0.5f * v * (1.0f + erff(v * 0.70710678118654752f));
            if (RES)      v += res[(size_t)grow * N + gcol];
            if (OUT_BF16) outB[(size_t)grow * N + gcol] = f2bf(v);
            else          outF[(size_t)grow * N + gcol] = v;
        }
    }
}

// ---------------------------------------------------------------------------
// Fused attention: one workgroup per (qblock(64), head, batch).
// The whole 64x1024 score strip lives in LDS as bf16 (129KB; CDNA5 has 320KB
// per WGP) -> zero HBM traffic for S.  QK^T (wmma) -> +rpe -> softmax -> PV.
// 256 threads = 8 waves: wave&3 picks the 16-row tile, wave>>2 splits the
// key-tile (phase 1) / dim-chunk (phase 4) work.
// ---------------------------------------------------------------------------
__global__ __launch_bounds__(256)
void attn_kernel(const unsigned short* __restrict__ qkv,   // [B*N, 2304] bf16
                 const float* __restrict__ rpe,            // [H, N, N] f32
                 unsigned short* __restrict__ o) {         // [B*N, 768] bf16
    constexpr int SSP = 1032;   // score pitch (ushort), padded
    constexpr int QP  = 72;     // 16B-aligned rows for async staging
    constexpr int KVP = 72;
    __shared__ unsigned short sS[64 * SSP];   // 129 KB (bf16 scores/probs)
    __shared__ unsigned short sQ[64 * QP];    // 9 KB
    __shared__ unsigned short sKV[64 * KVP];  // 9 KB (K then V staging)

    const int qb = blockIdx.x;        // 0..15
    const int h  = blockIdx.y;        // 0..11
    const int b  = blockIdx.z;        // 0..7
    const int tid  = threadIdx.x;
    const int wave = tid >> 5;
    const int lane = tid & 31;
    const int n0   = qb * 64;

    // ---- stage Q block 64x64 (512 x 16B chunks) ----
    for (int i = tid; i < 512; i += 256) {
        int r = i >> 3, c = (i & 7) << 3;
        copy16(&sQ[r * QP + c],
               &qkv[((size_t)b * SEQ + n0 + r) * 2304 + h * HDIM + c]);
    }
    async_wait0();
    __syncthreads();

    const int rw    = ((wave & 3) << 4) + (lane & 15);
    const int kbase = (lane >> 4) << 3;
    const int koff  = (lane >> 4) << 4;
    const int nn    = lane & 15;

    // persistent Q fragments (K-dim 64 = 2 steps of 32)
    v16u qfrag[2];
#pragma unroll
    for (int ks = 0; ks < 2; ++ks)
#pragma unroll
        for (int e = 0; e < 8; ++e) {
            qfrag[ks][e]     = sQ[rw * QP + ks * 32 + kbase + e];
            qfrag[ks][e + 8] = sQ[rw * QP + ks * 32 + kbase + 16 + e];
        }

    const float scale = 0.125f;   // HEAD_DIM^-0.5
    const int   ktsel = wave >> 2;
    const int   rloc  = ((wave & 3) << 4) + ((lane >> 4) << 3);

    // ---- phase 1: S = scale * Q K^T  (64 keys staged per iteration) ----
    for (int kbk = 0; kbk < 16; ++kbk) {
        __syncthreads();   // protect sKV reuse
        for (int i = tid; i < 512; i += 256) {
            int r = i >> 3, c = (i & 7) << 3;
            copy16(&sKV[r * KVP + c],
                   &qkv[((size_t)b * SEQ + kbk * 64 + r) * 2304 + DIMC + h * HDIM + c]);
        }
        async_wait0();
        __syncthreads();
#pragma unroll
        for (int t = 0; t < 2; ++t) {
            const int kt = ktsel * 2 + t;       // key tile 0..3 within 64
            v8f acc = {};
#pragma unroll
            for (int ks = 0; ks < 2; ++ks) {
                v16u bfrag;                      // B[d, key] = K[key, d]
#pragma unroll
                for (int e = 0; e < 16; ++e)
                    bfrag[e] = sKV[(kt * 16 + nn) * KVP + ks * 32 + kbase + e + ((e >> 3) << 3)];
                acc = wmma_bf16(qfrag[ks], bfrag, acc);
            }
#pragma unroll
            for (int i2 = 0; i2 < 8; ++i2)
                sS[(rloc + i2) * SSP + kbk * 64 + kt * 16 + (lane & 15)] =
                    f2bf(acc[i2] * scale);
        }
    }
    __syncthreads();

    // ---- phase 2: add relative position bias (coalesced f32 reads) ----
    const float* rrow = rpe + ((size_t)h * SEQ + n0) * SEQ;
    for (int i = tid; i < 64 * SEQ; i += 256) {
        int r = i >> 10, j = i & 1023;
        sS[r * SSP + j] = f2bf(bf2f(sS[r * SSP + j]) + rrow[(size_t)r * SEQ + j]);
    }
    __syncthreads();

    // ---- phase 3: softmax per row (one wave owns a row; wave32 shfl reduce) ----
    for (int r = wave; r < 64; r += 8) {
        float m = -3.0e38f;
        for (int j = lane; j < SEQ; j += 32) m = fmaxf(m, bf2f(sS[r * SSP + j]));
#pragma unroll
        for (int off = 16; off > 0; off >>= 1) m = fmaxf(m, __shfl_xor(m, off, 32));
        float s = 0.f;
        for (int j = lane; j < SEQ; j += 32) {
            float e = __expf(bf2f(sS[r * SSP + j]) - m);
            s += e;
            sS[r * SSP + j] = f2bf(e);
        }
#pragma unroll
        for (int off = 16; off > 0; off >>= 1) s += __shfl_xor(s, off, 32);
        const float inv = 1.0f / s;
        for (int j = lane; j < SEQ; j += 32)
            sS[r * SSP + j] = f2bf(bf2f(sS[r * SSP + j]) * inv);
    }
    __syncthreads();

    // ---- phase 4: O = P V  (K-dim 1024, 32 keys staged per iteration) ----
    const int dsel = wave >> 2;
    v8f oacc[2] = {};
    for (int kb2 = 0; kb2 < 32; ++kb2) {
        __syncthreads();
        {
            int i = tid;   // exactly 256 x 16B chunks
            int r = i >> 3, c = (i & 7) << 3;
            copy16(&sKV[r * KVP + c],
                   &qkv[((size_t)b * SEQ + kb2 * 32 + r) * 2304 + 2 * DIMC + h * HDIM + c]);
        }
        async_wait0();
        __syncthreads();
        v16u pfrag;
#pragma unroll
        for (int e = 0; e < 8; ++e) {
            pfrag[e]     = sS[rw * SSP + kb2 * 32 + kbase + e];
            pfrag[e + 8] = sS[rw * SSP + kb2 * 32 + kbase + 16 + e];
        }
#pragma unroll
        for (int t = 0; t < 2; ++t) {
            const int dc = dsel * 2 + t;        // output dim-chunk 0..3
            v16u vfrag;                          // B[key, d]
#pragma unroll
            for (int e = 0; e < 16; ++e)
                vfrag[e] = sKV[(kbase + e + ((e >> 3) << 3)) * KVP + dc * 16 + (lane & 15)];
            oacc[t] = wmma_bf16(pfrag, vfrag, oacc[t]);
        }
    }

    // ---- write O in [B*N, H*D] layout so proj GEMM consumes it directly ----
#pragma unroll
    for (int t = 0; t < 2; ++t) {
        const int dc = dsel * 2 + t;
#pragma unroll
        for (int i2 = 0; i2 < 8; ++i2) {
            const int n = n0 + rloc + i2;
            o[((size_t)b * SEQ + n) * DIMC + h * HDIM + dc * 16 + (lane & 15)] =
                f2bf(oacc[t][i2]);
        }
    }
}

// ---------------------------------------------------------------------------
// Host launch
// ---------------------------------------------------------------------------
extern "C" void kernel_launch(void* const* d_in, const int* in_sizes, int n_in,
                              void* d_out, int out_size, void* d_ws, size_t ws_size,
                              hipStream_t stream) {
    const float* x      = (const float*)d_in[0];
    const float* rpe    = (const float*)d_in[1];
    const float* qkv_w  = (const float*)d_in[2];
    const float* proj_w = (const float*)d_in[3];
    const float* proj_b = (const float*)d_in[4];
    const float* ln1_g  = (const float*)d_in[5];
    const float* ln1_b  = (const float*)d_in[6];
    const float* ln2_g  = (const float*)d_in[7];
    const float* ln2_b  = (const float*)d_in[8];
    const float* fc1_w  = (const float*)d_in[9];
    const float* fc1_b  = (const float*)d_in[10];
    const float* fc2_w  = (const float*)d_in[11];
    const float* fc2_b  = (const float*)d_in[12];
    float* out = (float*)d_out;

    char* ws = (char*)d_ws;
    size_t off = 0;
    auto alloc = [&](size_t bytes) -> void* {
        void* p = ws + off;
        off = (off + bytes + 255) & ~(size_t)255;
        return p;
    };
    unsigned short* wqkv  = (unsigned short*)alloc((size_t)DIMC * 3 * DIMC * 2);
    unsigned short* wproj = (unsigned short*)alloc((size_t)DIMC * DIMC * 2);
    unsigned short* wfc1  = (unsigned short*)alloc((size_t)DIMC * HIDDEN * 2);
    unsigned short* wfc2  = (unsigned short*)alloc((size_t)HIDDEN * DIMC * 2);
    unsigned short* hbuf  = (unsigned short*)alloc((size_t)MROWS * DIMC * 2);
    unsigned short* qkvb  = (unsigned short*)alloc((size_t)MROWS * 3 * DIMC * 2);
    unsigned short* obuf  = (unsigned short*)alloc((size_t)MROWS * DIMC * 2);
    float*          x1    = (float*)alloc((size_t)MROWS * DIMC * 4);
    unsigned short* ubuf  = (unsigned short*)alloc((size_t)MROWS * HIDDEN * 2);

    // weight conversion f32 -> bf16
    {
        int n;
        n = DIMC * 3 * DIMC;  cvt_f32_bf16<<<(n + 255) / 256, 256, 0, stream>>>(qkv_w,  wqkv,  n);
        n = DIMC * DIMC;      cvt_f32_bf16<<<(n + 255) / 256, 256, 0, stream>>>(proj_w, wproj, n);
        n = DIMC * HIDDEN;    cvt_f32_bf16<<<(n + 255) / 256, 256, 0, stream>>>(fc1_w,  wfc1,  n);
        n = HIDDEN * DIMC;    cvt_f32_bf16<<<(n + 255) / 256, 256, 0, stream>>>(fc2_w,  wfc2,  n);
    }

    // LN1 -> h
    ln_kernel<<<MROWS, 256, 0, stream>>>(x, ln1_g, ln1_b, hbuf);

    // qkv = h @ qkv_w          [8192,768] x [768,2304] -> bf16
    gemm_bf16_kernel<false, false, false, true>
        <<<dim3(MROWS / 128, 3 * DIMC / 64), 256, 0, stream>>>(
            hbuf, wqkv, nullptr, nullptr, nullptr, qkvb, MROWS, 3 * DIMC, DIMC);

    // fused attention -> o (bf16)
    attn_kernel<<<dim3(SEQ / 64, NHEADS, BATCH), 256, 0, stream>>>(qkvb, rpe, obuf);

    // x1 = x + o @ proj_w + proj_b          -> f32
    gemm_bf16_kernel<true, true, false, false>
        <<<dim3(MROWS / 128, DIMC / 64), 256, 0, stream>>>(
            obuf, wproj, proj_b, x, x1, nullptr, MROWS, DIMC, DIMC);

    // LN2 -> h (reuse)
    ln_kernel<<<MROWS, 256, 0, stream>>>(x1, ln2_g, ln2_b, hbuf);

    // u = gelu(h @ fc1_w + fc1_b)           -> bf16
    gemm_bf16_kernel<true, false, true, true>
        <<<dim3(MROWS / 128, HIDDEN / 64), 256, 0, stream>>>(
            hbuf, wfc1, fc1_b, nullptr, nullptr, ubuf, MROWS, HIDDEN, DIMC);

    // out = x1 + u @ fc2_w + fc2_b          -> f32
    gemm_bf16_kernel<true, true, false, false>
        <<<dim3(MROWS / 128, DIMC / 64), 256, 0, stream>>>(
            ubuf, wfc2, fc2_b, x1, out, nullptr, MROWS, DIMC, HIDDEN);
}
